// Point_MaDi_3186865734030
// MI455X (gfx1250) — matrix operand order repaired and checked by reading, not compiled, not run on hardware
//
#include <hip/hip_runtime.h>

// ---------------------------------------------------------------------------
// Point-MAE style network for MI455X (gfx1250, wave32, WMMA).
// All large GEMMs use v_wmma_f32_16x16x32_f16 (f16 in, f32 accumulate) with
// double-buffered LDS staging (one barrier per K-chunk) and global_prefetch_b8
// lookahead. BatchNorm statistics use deterministic fixed-order partial
// reductions (no float atomics). Activations kept in f16 to halve HBM traffic.
// ---------------------------------------------------------------------------

typedef __attribute__((ext_vector_type(16))) _Float16 v16h;
typedef __attribute__((ext_vector_type(8)))  float    v8f;
typedef __attribute__((ext_vector_type(2)))  _Float16 h2v;

#define B_  64
#define N_  2048
#define G_  64
#define M_  32
#define D_  384
#define H_  6
#define DEPTH_ 12
#define HID_ 1536
#define NTOK 4096          // B*G
#define NROW 131072        // B*G*M

__device__ __forceinline__ float gelu_f(float x) {
  return 0.5f * x * (1.0f + erff(x * 0.70710678118654752440f));
}

__device__ __forceinline__ v8f wmma16(v16h a, v16h b, v8f c) {
  // emits v_wmma_f32_16x16x32_f16
  return __builtin_amdgcn_wmma_f32_16x16x32_f16(false, a, false, b, (short)0, c,
                                                false, false);
}

// ---------------------------------------------------------------------------
// 1) Farthest point sampling: one block per batch, serial over G steps.
//    Matches the reference scan: fidx[g] = incoming "far", then update dists,
//    argmax (ties -> lowest index) becomes next "far".
// ---------------------------------------------------------------------------
__global__ __launch_bounds__(256) void fps_kernel(const float* __restrict__ xyz,
                                                  int* __restrict__ fidx,
                                                  float* __restrict__ center) {
  __shared__ float dist[N_];
  __shared__ float rv[256];
  __shared__ int   ri[256];
  __shared__ int   s_cur;
  const int b = blockIdx.x, t = threadIdx.x;
  const float* P = xyz + (size_t)b * N_ * 3;
  for (int i = t; i < N_; i += 256) dist[i] = 1e10f;
  if (t == 0) s_cur = 0;
  __syncthreads();
  for (int g = 0; g < G_; ++g) {
    const int cur = s_cur;
    const float px = P[cur * 3 + 0], py = P[cur * 3 + 1], pz = P[cur * 3 + 2];
    if (t == 0) {
      fidx[b * G_ + g] = cur;
      center[(size_t)(b * G_ + g) * 3 + 0] = px;
      center[(size_t)(b * G_ + g) * 3 + 1] = py;
      center[(size_t)(b * G_ + g) * 3 + 2] = pz;
    }
    float bestv = -1.0f; int besti = 0;
    for (int i = t; i < N_; i += 256) {
      const float dx = P[i * 3 + 0] - px;
      const float dy = P[i * 3 + 1] - py;
      const float dz = P[i * 3 + 2] - pz;
      const float d  = dx * dx + dy * dy + dz * dz;
      const float nd = fminf(dist[i], d);
      dist[i] = nd;
      if (nd > bestv) { bestv = nd; besti = i; }
    }
    rv[t] = bestv; ri[t] = besti;
    __syncthreads();
    for (int s = 128; s > 0; s >>= 1) {
      if (t < s) {
        if (rv[t + s] > rv[t] || (rv[t + s] == rv[t] && ri[t + s] < ri[t])) {
          rv[t] = rv[t + s]; ri[t] = ri[t + s];
        }
      }
      __syncthreads();
    }
    if (t == 0) s_cur = ri[0];
    __syncthreads();
  }
}

// ---------------------------------------------------------------------------
// 2) kNN (top-M smallest d2, stable tie order) + relative coords.
//    d2 uses the same |c|^2+|p|^2-2c.p formula as the reference.
// ---------------------------------------------------------------------------
__global__ __launch_bounds__(256) void knn_kernel(const float* __restrict__ xyz,
                                                  const float* __restrict__ center,
                                                  float* __restrict__ nbr) {
  __shared__ float d2[N_];
  __shared__ float rv[256];
  __shared__ int   ri[256];
  const int bg = blockIdx.x, b = bg >> 6, t = threadIdx.x;
  const float* P = xyz + (size_t)b * N_ * 3;
  const float cx = center[(size_t)bg * 3 + 0];
  const float cy = center[(size_t)bg * 3 + 1];
  const float cz = center[(size_t)bg * 3 + 2];
  const float c2 = cx * cx + cy * cy + cz * cz;
  for (int i = t; i < N_; i += 256) {
    const float x = P[i * 3 + 0], y = P[i * 3 + 1], z = P[i * 3 + 2];
    d2[i] = c2 + (x * x + y * y + z * z) - 2.0f * (cx * x + cy * y + cz * z);
  }
  __syncthreads();
  for (int m = 0; m < M_; ++m) {
    float bv = 1e30f; int bi = 0;
    for (int i = t; i < N_; i += 256) {
      const float d = d2[i];
      if (d < bv) { bv = d; bi = i; }
    }
    rv[t] = bv; ri[t] = bi;
    __syncthreads();
    for (int s = 128; s > 0; s >>= 1) {
      if (t < s) {
        if (rv[t + s] < rv[t] || (rv[t + s] == rv[t] && ri[t + s] < ri[t])) {
          rv[t] = rv[t + s]; ri[t] = ri[t + s];
        }
      }
      __syncthreads();
    }
    if (t == 0) {
      const int idx = ri[0];
      d2[idx] = 1e30f;
      float* o = nbr + ((size_t)bg * M_ + m) * 3;
      o[0] = P[idx * 3 + 0] - cx;
      o[1] = P[idx * 3 + 1] - cy;
      o[2] = P[idx * 3 + 2] - cz;
    }
    __syncthreads();
  }
}

// ---------------------------------------------------------------------------
// 3) BN1 statistics: t1 = nbr @ w1a + b1a (K=3, VALU). Deterministic
//    per-block partial sums.
// ---------------------------------------------------------------------------
__global__ __launch_bounds__(128) void stats1_kernel(const float* __restrict__ nbr,
                                                     const float* __restrict__ w1a,
                                                     const float* __restrict__ b1a,
                                                     float* __restrict__ ps,
                                                     float* __restrict__ pq) {
  const int c = threadIdx.x;          // channel 0..127
  const int r0 = blockIdx.x * 256;    // 512 blocks * 256 rows = 131072
  const float w0 = w1a[c], w1 = w1a[128 + c], w2 = w1a[256 + c], bb = b1a[c];
  float s = 0.0f, q = 0.0f;
  for (int r = 0; r < 256; ++r) {
    const float* p = nbr + (size_t)(r0 + r) * 3;
    const float tv = p[0] * w0 + p[1] * w1 + p[2] * w2 + bb;
    s += tv; q += tv * tv;
  }
  ps[(size_t)blockIdx.x * 128 + c] = s;
  pq[(size_t)blockIdx.x * 128 + c] = q;
}

// reduce partials (fixed order) and fold BN into scale/shift
__global__ void bn_finalize(const float* __restrict__ ps, const float* __restrict__ pq,
                            int nparts, int C, float inv_cnt,
                            const float* __restrict__ g, const float* __restrict__ b,
                            float* __restrict__ scale, float* __restrict__ shift) {
  const int c = blockIdx.x * blockDim.x + threadIdx.x;
  if (c >= C) return;
  float s = 0.0f, q = 0.0f;
  for (int i = 0; i < nparts; ++i) { s += ps[(size_t)i * C + c]; q += pq[(size_t)i * C + c]; }
  const float m  = s * inv_cnt;
  const float v  = q * inv_cnt - m * m;
  const float sc = g[c] * rsqrtf(v + 1e-5f);
  scale[c] = sc;
  shift[c] = b[c] - m * sc;
}

// a1 = relu(bn1(nbr @ w1a + b1a)) stored f16
__global__ __launch_bounds__(256) void a1_kernel(const float* __restrict__ nbr,
                                                 const float* __restrict__ w1a,
                                                 const float* __restrict__ b1a,
                                                 const float* __restrict__ scale,
                                                 const float* __restrict__ shift,
                                                 _Float16* __restrict__ a1) {
  const int idx = blockIdx.x * 256 + threadIdx.x;   // 131072*128 elements
  const int row = idx >> 7, c = idx & 127;
  const float* p = nbr + (size_t)row * 3;
  float tv = p[0] * w1a[c] + p[1] * w1a[128 + c] + p[2] * w1a[256 + c] + b1a[c];
  tv = tv * scale[c] + shift[c];
  a1[(size_t)idx] = (_Float16)fmaxf(tv, 0.0f);
}

// ---------------------------------------------------------------------------
// Generic WMMA GEMM: D[M,N] = epilogue(A[M,K] @ W[K,N]).
//   AT     : A element type (float or _Float16), converted to f16 in LDS
//   ACT    : 0 none, 1 relu, 2 exact gelu
//   OUT_F16: store f16 instead of f32
//   RESID  : 0 none, 1 per-row resid[M,N], 2 per-group resid[M/32,N]
//   STATS  : write deterministic per-block partial sums/sumsq instead of D
//   SCALE  : y = y*scale[n]+shift[n] (folded BN) before activation
// Block: 256 threads (8 waves), 64x64 output tile, K chunks of 32.
// Double-buffered LDS (one barrier per chunk) + global_prefetch_b8 lookahead.
// WMMA lane layouts per ISA 7.12.2 (A and B: k=2(v&3)+16(v>>2)+8h).
// ---------------------------------------------------------------------------
template <typename AT, int ACT, bool OUT_F16, int RESID, bool STATS, bool SCALE>
__global__ __launch_bounds__(256)
void gemm_wmma_kernel(const AT* __restrict__ A, const float* __restrict__ W,
                      const float* __restrict__ bias, const float* __restrict__ resid,
                      const float* __restrict__ scale, const float* __restrict__ shift,
                      void* __restrict__ outp, float* __restrict__ s_sum,
                      float* __restrict__ s_sq, int M, int N, int K) {
  __shared__ _Float16 As[2][64][40];
  __shared__ _Float16 Bs[2][64][40];
  __shared__ float red_s[64][8];
  __shared__ float red_q[64][8];

  const int m0 = blockIdx.x * 64;
  const int n0 = blockIdx.y * 64;
  const int tid = threadIdx.x;
  const int w = tid >> 5, lane = tid & 31;
  const int half = lane >> 4, lr = lane & 15;
  const int mr = w & 3;            // m sub-tile 0..3
  const int nbase = w >> 2;        // n sub-tile base (0 or 1); also +2

  // staging assignments (coalesced 8-wide per thread)
  const int arow = tid >> 2, aseg = tid & 3;
  const int bk = tid >> 3, bn8 = tid & 7;
  const AT* aptr = A + (size_t)(m0 + arow) * K + aseg * 8;

  v8f c0 = {};
  v8f c1 = {};

  // prologue: stage chunk 0 into buffer 0
  {
#pragma unroll
    for (int j = 0; j < 8; ++j) As[0][arow][aseg * 8 + j] = (_Float16)aptr[j];
    const float* wr = W + (size_t)bk * N + n0 + bn8 * 8;
#pragma unroll
    for (int j = 0; j < 8; ++j) Bs[0][bn8 * 8 + j][bk] = (_Float16)wr[j];
  }
  __syncthreads();

  const int NC = K >> 5;
  for (int c = 0; c < NC; ++c) {
    const int cur = c & 1;
    const bool more = (c + 1 < NC);
    AT    areg[8];
    float breg[8];
    if (more) {   // issue next-chunk global loads (hidden behind WMMAs)
      const AT* ap = aptr + (size_t)(c + 1) * 32;
#pragma unroll
      for (int j = 0; j < 8; ++j) areg[j] = ap[j];
      const float* wr = W + (size_t)((c + 1) * 32 + bk) * N + n0 + bn8 * 8;
#pragma unroll
      for (int j = 0; j < 8; ++j) breg[j] = wr[j];
      if (c + 2 < NC) {   // global_prefetch_b8 lookahead for chunk c+2
        __builtin_prefetch(aptr + (size_t)(c + 2) * 32, 0, 3);
        __builtin_prefetch(W + (size_t)((c + 2) * 32 + bk) * N + n0 + bn8 * 8, 0, 3);
      }
    }

    // fragment gather + WMMA from current buffer
    v16h a, b0, b1;
#pragma unroll
    for (int v = 0; v < 8; ++v) {
      const int ka = 2 * (v & 3) + 16 * (v >> 2) + 8 * half;  // A layout
      const int kb = 2 * (v & 3) + 16 * (v >> 2) + 8 * half;  // B layout
      h2v av  = *(const h2v*)&As[cur][16 * mr + lr][ka];
      h2v bv0 = *(const h2v*)&Bs[cur][16 * nbase + lr][kb];
      h2v bv1 = *(const h2v*)&Bs[cur][16 * (nbase + 2) + lr][kb];
      a[2 * v] = av[0];  a[2 * v + 1] = av[1];
      b0[2 * v] = bv0[0]; b0[2 * v + 1] = bv0[1];
      b1[2 * v] = bv1[0]; b1[2 * v + 1] = bv1[1];
    }
    c0 = wmma16(a, b0, c0);
    c1 = wmma16(a, b1, c1);

    if (more) {   // commit next chunk to the other buffer
      const int nxt = cur ^ 1;
#pragma unroll
      for (int j = 0; j < 8; ++j) As[nxt][arow][aseg * 8 + j] = (_Float16)areg[j];
#pragma unroll
      for (int j = 0; j < 8; ++j) Bs[nxt][bn8 * 8 + j][bk] = (_Float16)breg[j];
    }
    __syncthreads();
  }

  // epilogue
#pragma unroll
  for (int ht = 0; ht < 2; ++ht) {
    v8f cc = ht ? c1 : c0;
    const int nl = 16 * (nbase + 2 * ht) + lr;   // 0..63
    const int n  = n0 + nl;
    const float bsv = bias ? bias[n] : 0.0f;
    float ssum = 0.0f, ssq = 0.0f;
#pragma unroll
    for (int r = 0; r < 8; ++r) {
      const int m = m0 + 16 * mr + r + 8 * half;
      float val = cc[r] + bsv;
      if constexpr (RESID == 1) val += resid[(size_t)m * N + n];
      if constexpr (RESID == 2) val += resid[(size_t)(m >> 5) * N + n];
      if constexpr (SCALE) val = val * scale[n] + shift[n];
      if constexpr (ACT == 1) val = fmaxf(val, 0.0f);
      if constexpr (ACT == 2) val = gelu_f(val);
      if constexpr (STATS) {
        ssum += val; ssq += val * val;
      } else {
        if constexpr (OUT_F16) ((_Float16*)outp)[(size_t)m * N + n] = (_Float16)val;
        else                   ((float*)outp)[(size_t)m * N + n]    = val;
      }
    }
    if constexpr (STATS) {
      red_s[nl][mr * 2 + half] = ssum;
      red_q[nl][mr * 2 + half] = ssq;
    }
  }
  if constexpr (STATS) {
    __syncthreads();
    if (tid < 64) {
      float s = 0.0f, q = 0.0f;
#pragma unroll
      for (int i = 0; i < 8; ++i) { s += red_s[tid][i]; q += red_q[tid][i]; }
      s_sum[(size_t)blockIdx.x * N + n0 + tid] = s;
      s_sq [(size_t)blockIdx.x * N + n0 + tid] = q;
    }
  }
}

// ---------------------------------------------------------------------------
// per-group channel max of f1 -> fg (f16)
// ---------------------------------------------------------------------------
__global__ __launch_bounds__(256) void fgmax_kernel(const _Float16* __restrict__ f1,
                                                    _Float16* __restrict__ fg) {
  const int bg = blockIdx.x, c = threadIdx.x;   // C=256
  float mv = -1e30f;
  for (int m = 0; m < M_; ++m)
    mv = fmaxf(mv, (float)f1[((size_t)bg * M_ + m) * 256 + c]);
  fg[(size_t)bg * 256 + c] = (_Float16)mv;
}

// ---------------------------------------------------------------------------
// Fused: out[bg,:] = max_m( a2[bg,32,512] @ w2b[512,384] ) + b2b
// One block per group; a2 tile lives in LDS; cross-half max via shfl_xor 16.
// ---------------------------------------------------------------------------
__global__ __launch_bounds__(256) void pnmax_kernel(const _Float16* __restrict__ a2,
                                                    const float* __restrict__ w2b,
                                                    const float* __restrict__ b2b,
                                                    float* __restrict__ x) {
  __shared__ _Float16 As[32][520];   // 32 x 512 (+8 pad)
  __shared__ _Float16 Bs[384][36];   // transposed w2b chunk
  const int bg = blockIdx.x;
  const int tid = threadIdx.x;
  const int w = tid >> 5, lane = tid & 31, half = lane >> 4, lr = lane & 15;

  const _Float16* ag = a2 + (size_t)bg * M_ * 512;
  for (int e = tid; e < 32 * 512; e += 256) As[e >> 9][e & 511] = ag[e];

  v8f vz = {};
  v8f acc0[3], acc1[3];
#pragma unroll
  for (int j = 0; j < 3; ++j) { acc0[j] = vz; acc1[j] = vz; }

  for (int k0 = 0; k0 < 512; k0 += 32) {
    for (int e = tid; e < 384 * 32; e += 256) {
      const int n = e >> 5, kk = e & 31;
      Bs[n][kk] = (_Float16)w2b[(size_t)(k0 + kk) * D_ + n];
    }
    if (k0 + 32 < 512)   // global_prefetch_b8 lookahead for next w2b chunk
      __builtin_prefetch(w2b + (size_t)(k0 + 32) * D_ + (size_t)tid * 48, 0, 3);
    __syncthreads();
    v16h a0, a1;
#pragma unroll
    for (int v = 0; v < 8; ++v) {
      const int ka = 2 * (v & 3) + 16 * (v >> 2) + 8 * half + k0;
      h2v t0 = *(const h2v*)&As[lr][ka];
      h2v t1 = *(const h2v*)&As[16 + lr][ka];
      a0[2 * v] = t0[0]; a0[2 * v + 1] = t0[1];
      a1[2 * v] = t1[0]; a1[2 * v + 1] = t1[1];
    }
#pragma unroll
    for (int j = 0; j < 3; ++j) {
      const int ns = w + 8 * j;
      v16h bf;
#pragma unroll
      for (int v = 0; v < 8; ++v) {
        const int kb = 2 * (v & 3) + 16 * (v >> 2) + 8 * half;
        h2v bv = *(const h2v*)&Bs[16 * ns + lr][kb];
        bf[2 * v] = bv[0]; bf[2 * v + 1] = bv[1];
      }
      acc0[j] = wmma16(a0, bf, acc0[j]);
      acc1[j] = wmma16(a1, bf, acc1[j]);
    }
    __syncthreads();
  }
#pragma unroll
  for (int j = 0; j < 3; ++j) {
    const int n = 16 * (w + 8 * j) + lr;   // 0..383
    float mv = -1e30f;
#pragma unroll
    for (int r = 0; r < 8; ++r) {
      mv = fmaxf(mv, acc0[j][r]);
      mv = fmaxf(mv, acc1[j][r]);
    }
    mv = fmaxf(mv, __shfl_xor(mv, 16, 32));
    if (half == 0) x[(size_t)bg * D_ + n] = mv + b2b[n];
  }
}

// ---------------------------------------------------------------------------
// Positional embedding: gelu(center @ pw1 + pb1) @ pw2 + pb2
// ---------------------------------------------------------------------------
__global__ __launch_bounds__(128) void pos_kernel(const float* __restrict__ center,
                                                  const float* __restrict__ pw1,
                                                  const float* __restrict__ pb1,
                                                  const float* __restrict__ pw2,
                                                  const float* __restrict__ pb2,
                                                  float* __restrict__ pos) {
  __shared__ float t[128];
  const int bg = blockIdx.x, tid = threadIdx.x;
  const float cx = center[(size_t)bg * 3 + 0];
  const float cy = center[(size_t)bg * 3 + 1];
  const float cz = center[(size_t)bg * 3 + 2];
  const float u = cx * pw1[tid] + cy * pw1[128 + tid] + cz * pw1[256 + tid] + pb1[tid];
  t[tid] = gelu_f(u);
  __syncthreads();
  for (int n = tid; n < D_; n += 128) {
    float s = pb2[n];
    for (int c = 0; c < 128; ++c) s += t[c] * pw2[(size_t)c * D_ + n];
    pos[(size_t)bg * D_ + n] = s;
  }
}

// ---------------------------------------------------------------------------
// LayerNorm over D=384 (one block per token). ADDPOS: x += pos (written back).
// ---------------------------------------------------------------------------
template <bool ADDPOS>
__global__ __launch_bounds__(128) void ln_kernel(float* __restrict__ x,
                                                 const float* __restrict__ pos,
                                                 const float* __restrict__ g,
                                                 const float* __restrict__ b,
                                                 float* __restrict__ out) {
  __shared__ float red[128];
  const int tok = blockIdx.x, t = threadIdx.x;
  float* xr = x + (size_t)tok * D_;
  float v0 = xr[t], v1 = xr[t + 128], v2 = xr[t + 256];
  if constexpr (ADDPOS) {
    const float* pr = pos + (size_t)tok * D_;
    v0 += pr[t]; v1 += pr[t + 128]; v2 += pr[t + 256];
    xr[t] = v0; xr[t + 128] = v1; xr[t + 256] = v2;
  }
  red[t] = v0 + v1 + v2;
  __syncthreads();
  for (int s = 64; s > 0; s >>= 1) { if (t < s) red[t] += red[t + s]; __syncthreads(); }
  const float mean = red[0] * (1.0f / 384.0f);
  __syncthreads();
  const float d0 = v0 - mean, d1 = v1 - mean, d2 = v2 - mean;
  red[t] = d0 * d0 + d1 * d1 + d2 * d2;
  __syncthreads();
  for (int s = 64; s > 0; s >>= 1) { if (t < s) red[t] += red[t + s]; __syncthreads(); }
  const float inv = rsqrtf(red[0] * (1.0f / 384.0f) + 1e-5f);
  float* o = out + (size_t)tok * D_;
  o[t]       = d0 * inv * g[t]       + b[t];
  o[t + 128] = d1 * inv * g[t + 128] + b[t + 128];
  o[t + 256] = d2 * inv * g[t + 256] + b[t + 256];
}

// ---------------------------------------------------------------------------
// Attention: one block per (batch, head). G=64, dh=64 -> tiny, VALU is fine
// (~1.6% of total FLOPs; WMMA would not move the roofline here).
// ---------------------------------------------------------------------------
__global__ __launch_bounds__(256) void attn_kernel(const float* __restrict__ qkv,
                                                   float* __restrict__ attno) {
  __shared__ float QV[64][64];   // Q, later reused for V
  __shared__ float Kb[64][64];
  __shared__ float S[64][65];
  const int bh = blockIdx.x, b = bh / H_, h = bh % H_;
  const int t = threadIdx.x;
  const float* base = qkv + (size_t)(b * G_) * (3 * D_) + h * 64;
  for (int e = t; e < 4096; e += 256) {
    const int g = e >> 6, d = e & 63;
    const float* r = base + (size_t)g * (3 * D_);
    QV[g][d] = r[d];
    Kb[g][d] = r[D_ + d];
  }
  __syncthreads();
  for (int e = t; e < 4096; e += 256) {
    const int i = e >> 6, j = e & 63;
    float s = 0.0f;
    for (int d = 0; d < 64; ++d) s += QV[i][d] * Kb[j][d];
    S[i][j] = s * 0.125f;   // dh^-0.5
  }
  __syncthreads();
  { // softmax: 4 threads per row
    const int row = t >> 2, sub = t & 3;
    float mx = -1e30f;
    for (int j = sub; j < 64; j += 4) mx = fmaxf(mx, S[row][j]);
    mx = fmaxf(mx, __shfl_xor(mx, 1, 4));
    mx = fmaxf(mx, __shfl_xor(mx, 2, 4));
    float sum = 0.0f;
    for (int j = sub; j < 64; j += 4) { const float e2 = expf(S[row][j] - mx); S[row][j] = e2; sum += e2; }
    sum += __shfl_xor(sum, 1, 4);
    sum += __shfl_xor(sum, 2, 4);
    const float inv = 1.0f / sum;
    for (int j = sub; j < 64; j += 4) S[row][j] *= inv;
  }
  // load V into QV (Q is dead; S reads finished before overwrite via barrier)
  for (int e = t; e < 4096; e += 256) {
    const int g = e >> 6, d = e & 63;
    QV[g][d] = base[(size_t)g * (3 * D_) + 2 * D_ + d];
  }
  __syncthreads();
  float* ob = attno + (size_t)(b * G_) * D_ + h * 64;
  for (int e = t; e < 4096; e += 256) {
    const int i = e >> 6, j = e & 63;
    float s = 0.0f;
    for (int k = 0; k < 64; ++k) s += S[i][k] * QV[k][j];
    ob[(size_t)i * D_ + j] = s;
  }
}

// ---------------------------------------------------------------------------
// Host orchestration
// ---------------------------------------------------------------------------
extern "C" void kernel_launch(void* const* d_in, const int* in_sizes, int n_in,
                              void* d_out, int out_size, void* d_ws, size_t ws_size,
                              hipStream_t stream) {
  (void)in_sizes; (void)n_in; (void)out_size; (void)ws_size;
  const float* xyz   = (const float*)d_in[0];
  const float* w1a   = (const float*)d_in[1];
  const float* b1a   = (const float*)d_in[2];
  const float* bn1_g = (const float*)d_in[3];
  const float* bn1_b = (const float*)d_in[4];
  const float* w1b   = (const float*)d_in[5];
  const float* b1b   = (const float*)d_in[6];
  const float* w2a   = (const float*)d_in[7];
  const float* b2a   = (const float*)d_in[8];
  const float* bn2_g = (const float*)d_in[9];
  const float* bn2_b = (const float*)d_in[10];
  const float* w2b   = (const float*)d_in[11];
  const float* b2b   = (const float*)d_in[12];
  const float* pw1   = (const float*)d_in[13];
  const float* pb1   = (const float*)d_in[14];
  const float* pw2   = (const float*)d_in[15];
  const float* pb2   = (const float*)d_in[16];
  const float* ln1_g = (const float*)d_in[17];
  const float* ln1_b = (const float*)d_in[18];
  const float* qkv_w = (const float*)d_in[19];
  const float* proj_w= (const float*)d_in[20];
  const float* proj_b= (const float*)d_in[21];
  const float* ln2_g = (const float*)d_in[22];
  const float* ln2_b = (const float*)d_in[23];
  const float* fc1_w = (const float*)d_in[24];
  const float* fc1_b = (const float*)d_in[25];
  const float* fc2_w = (const float*)d_in[26];
  const float* fc2_b = (const float*)d_in[27];
  const float* norm_g= (const float*)d_in[28];
  const float* norm_b= (const float*)d_in[29];

  char* ws = (char*)d_ws;
  size_t off = 0;
  auto alloc = [&](size_t bytes) -> void* {
    void* p = ws + off;
    off += (bytes + 255) & ~(size_t)255;
    return p;
  };

  int*   fidx   = (int*)  alloc((size_t)NTOK * 4);
  float* center = (float*)alloc((size_t)NTOK * 3 * 4);
  float* nbr    = (float*)alloc((size_t)NROW * 3 * 4);
  float* p1s    = (float*)alloc((size_t)512 * 128 * 4);
  float* p1q    = (float*)alloc((size_t)512 * 128 * 4);
  float* scale1 = (float*)alloc(128 * 4);
  float* shift1 = (float*)alloc(128 * 4);
  float* p2s    = (float*)alloc((size_t)2048 * 512 * 4);
  float* p2q    = (float*)alloc((size_t)2048 * 512 * 4);
  float* scale2 = (float*)alloc(512 * 4);
  float* shift2 = (float*)alloc(512 * 4);
  _Float16* fg  = (_Float16*)alloc((size_t)NTOK * 256 * 2);
  float* gvec   = (float*)alloc((size_t)NTOK * 512 * 4);
  float* x      = (float*)alloc((size_t)NTOK * D_ * 4);
  float* pos    = (float*)alloc((size_t)NTOK * D_ * 4);
  // arenas (reused across phases)
  char* arenaA  = (char*)alloc((size_t)NROW * 128 * 2);   // a1 (33.5MB)
  char* arenaB  = (char*)alloc((size_t)NROW * 256 * 2);   // f1 (67MB)
  char* arenaC  = (char*)alloc((size_t)NROW * 512 * 2);   // a2 (134MB)

  _Float16* a1 = (_Float16*)arenaA;
  _Float16* f1 = (_Float16*)arenaB;
  _Float16* a2 = (_Float16*)arenaC;
  // transformer buffers alias the PointNet arenas (phases are disjoint)
  float* hbuf  = (float*)arenaA;                                      // 6.3MB
  float* qkvb  = (float*)(arenaA + (size_t)NTOK * D_ * 4);            // 18.9MB
  float* attno = (float*)(arenaA + (size_t)NTOK * D_ * 4 + (size_t)NTOK * 3 * D_ * 4);
  float* hid   = (float*)arenaB;                                      // 25.2MB

  // ---- grouping ----
  fps_kernel<<<B_, 256, 0, stream>>>(xyz, fidx, center);
  knn_kernel<<<NTOK, 256, 0, stream>>>(xyz, center, nbr);

  // ---- PointNet stage 1 ----
  stats1_kernel<<<512, 128, 0, stream>>>(nbr, w1a, b1a, p1s, p1q);
  bn_finalize<<<1, 128, 0, stream>>>(p1s, p1q, 512, 128, 1.0f / (float)NROW,
                                     bn1_g, bn1_b, scale1, shift1);
  a1_kernel<<<(NROW * 128) / 256, 256, 0, stream>>>(nbr, w1a, b1a, scale1, shift1, a1);
  // f1 = a1 @ w1b + b1b  (f16 out)
  gemm_wmma_kernel<_Float16, 0, true, 0, false, false>
      <<<dim3(NROW / 64, 256 / 64), 256, 0, stream>>>(
          a1, w1b, b1b, nullptr, nullptr, nullptr, f1, nullptr, nullptr,
          NROW, 256, 128);
  fgmax_kernel<<<NTOK, 256, 0, stream>>>(f1, fg);

  // ---- PointNet stage 2 ----
  // gvec = fg @ w2a[0:256] + b2a  (per-group vector, f32)
  gemm_wmma_kernel<_Float16, 0, false, 0, false, false>
      <<<dim3(NTOK / 64, 512 / 64), 256, 0, stream>>>(
          fg, w2a, b2a, nullptr, nullptr, nullptr, gvec, nullptr, nullptr,
          NTOK, 512, 256);
  // t2 stats: t2 = f1 @ w2a[256:512] + gvec[row/32]  (deterministic partials)
  gemm_wmma_kernel<_Float16, 0, false, 2, true, false>
      <<<dim3(NROW / 64, 512 / 64), 256, 0, stream>>>(
          f1, w2a + 256 * 512, nullptr, gvec, nullptr, nullptr, nullptr,
          p2s, p2q, NROW, 512, 256);
  bn_finalize<<<4, 128, 0, stream>>>(p2s, p2q, 2048, 512, 1.0f / (float)NROW,
                                     bn2_g, bn2_b, scale2, shift2);
  // a2 = relu(bn2(t2))  (f16 out; t2 recomputed — cheaper than storing it)
  gemm_wmma_kernel<_Float16, 1, true, 2, false, true>
      <<<dim3(NROW / 64, 512 / 64), 256, 0, stream>>>(
          f1, w2a + 256 * 512, nullptr, gvec, scale2, shift2, a2,
          nullptr, nullptr, NROW, 512, 256);
  // x[bg,:] = max_m(a2 @ w2b) + b2b   (fused WMMA + max)
  pnmax_kernel<<<NTOK, 256, 0, stream>>>(a2, w2b, b2b, x);

  // ---- positional embedding ----
  pos_kernel<<<NTOK, 128, 0, stream>>>(center, pw1, pb1, pw2, pb2, pos);

  // ---- transformer ----
  for (int i = 0; i < DEPTH_; ++i) {
    ln_kernel<true><<<NTOK, 128, 0, stream>>>(x, pos, ln1_g + i * D_, ln1_b + i * D_, hbuf);
    gemm_wmma_kernel<float, 0, false, 0, false, false>
        <<<dim3(NTOK / 64, (3 * D_) / 64), 256, 0, stream>>>(
            hbuf, qkv_w + (size_t)i * D_ * 3 * D_, nullptr, nullptr, nullptr,
            nullptr, qkvb, nullptr, nullptr, NTOK, 3 * D_, D_);
    attn_kernel<<<B_ * H_, 256, 0, stream>>>(qkvb, attno);
    gemm_wmma_kernel<float, 0, false, 1, false, false>
        <<<dim3(NTOK / 64, D_ / 64), 256, 0, stream>>>(
            attno, proj_w + (size_t)i * D_ * D_, proj_b + i * D_, x, nullptr,
            nullptr, x, nullptr, nullptr, NTOK, D_, D_);
    ln_kernel<false><<<NTOK, 128, 0, stream>>>(x, nullptr, ln2_g + i * D_, ln2_b + i * D_, hbuf);
    gemm_wmma_kernel<float, 2, false, 0, false, false>
        <<<dim3(NTOK / 64, HID_ / 64), 256, 0, stream>>>(
            hbuf, fc1_w + (size_t)i * D_ * HID_, fc1_b + i * HID_, nullptr,
            nullptr, nullptr, hid, nullptr, nullptr, NTOK, HID_, D_);
    gemm_wmma_kernel<float, 0, false, 1, false, false>
        <<<dim3(NTOK / 64, D_ / 64), 256, 0, stream>>>(
            hid, fc2_w + (size_t)i * HID_ * D_, fc2_b + i * D_, x, nullptr,
            nullptr, x, nullptr, nullptr, NTOK, D_, HID_);
  }

  // ---- final LayerNorm into output ----
  ln_kernel<false><<<NTOK, 128, 0, stream>>>(x, nullptr, norm_g, norm_b, (float*)d_out);
}
